// SchNetOnlyModel_34866544509062
// MI455X (gfx1250) — matrix operand order repaired and checked by reading, not compiled, not run on hardware
//
#include <hip/hip_runtime.h>

#define NATOMS   50000
#define HD       128
#define NGAUSS   10
#define NBATCH   64
#define NTGT     4
#define CUTOFF_F 10.0f

typedef __bf16 bf16_t;
typedef __bf16 v16bf __attribute__((ext_vector_type(16)));
typedef __bf16 v8bf  __attribute__((ext_vector_type(8)));
typedef float  v8f   __attribute__((ext_vector_type(8)));

__device__ __forceinline__ float sspf(float x) {
    // shifted softplus: log(1+exp(x)) - log(2)
    const float LN2 = 0.6931471805599453f;
    float sp = (x > 20.0f) ? x : log1pf(__expf(x));
    return sp - LN2;
}

// ---------------------------------------------------------------- utilities
__global__ void zero_kernel(float* p, long long n) {
    long long i = (long long)blockIdx.x * blockDim.x + threadIdx.x;
    if (i < n) p[i] = 0.0f;
}

__global__ void init_h_kernel(float* h, const float* emb, const int* z, long long nh) {
    long long i = (long long)blockIdx.x * blockDim.x + threadIdx.x;
    if (i < nh) {
        int n = (int)(i / HD);
        int c = (int)(i % HD);
        h[i] = emb[(long long)z[n] * HD + c];
    }
}

// ------------------------------------------------ generic WMMA node matmul
// Y[nrows,NCOLS] = op( X[nrows,K] @ W[K,NCOLS] + bias )
// MODE 0: store   MODE 1: Y += val (residual)   MODE 2: atomicAdd into Y[seg[row]]
// All shape params are template constants so the accumulator array fully
// scalarizes into VGPRs (no scratch spills) and LDS offsets are immediates.
template <int K, int NCOLS, int DO_SSP, int MODE>
__global__ __launch_bounds__(256) void wmma_matmul_kernel(
    const float* __restrict__ X, const float* __restrict__ W,
    const float* __restrict__ bias, float* __restrict__ Y,
    const int* __restrict__ seg, int nrows)
{
    constexpr int NTC = NCOLS / 16;   // # of 16-col output tiles
    constexpr int KKC = K / 32;       // # of K=32 WMMA steps

    __shared__ __align__(16) bf16_t Wt[NCOLS * K];   // W transposed, bf16: Wt[n*K+k]

    const int tid = threadIdx.x;
    // stage W (f32 global, [K][NCOLS] row-major) -> LDS transposed bf16
    for (int idx = tid; idx < K * NCOLS; idx += 256) {
        int k = idx / NCOLS;
        int n = idx % NCOLS;
        Wt[n * K + k] = (bf16_t)W[idx];
    }
    __syncthreads();

    const int lane   = tid & 31;
    const int wave   = tid >> 5;
    const int gwave  = blockIdx.x * 8 + wave;
    const int nwaves = gridDim.x * 8;
    const int tiles  = (nrows + 15) >> 4;
    const int m16    = lane & 15;
    const int hi     = lane >> 4;

    for (int t = gwave; t < tiles; t += nwaves) {
        const int rowBase = t << 4;
        const int myRow   = rowBase + m16;
        const bool rowOK  = myRow < nrows;

        v8f acc[NTC];
#pragma unroll
        for (int nt = 0; nt < NTC; ++nt)
#pragma unroll
            for (int r = 0; r < 8; ++r) acc[nt][r] = 0.0f;

#pragma unroll
        for (int kk = 0; kk < KKC; ++kk) {
            // A fragment: row = m16, K-cols = kk*32 + hi*8 + {0..7, 16..23}
            v16bf a;
#pragma unroll
            for (int j = 0; j < 16; ++j) a[j] = (bf16_t)0.0f;
            if (rowOK) {
                const float* xp = X + (long long)myRow * K + kk * 32 + hi * 8;
                float4 f0 = *(const float4*)(xp);
                float4 f1 = *(const float4*)(xp + 4);
                float4 f2 = *(const float4*)(xp + 16);
                float4 f3 = *(const float4*)(xp + 20);
                a[0] = (bf16_t)f0.x;  a[1] = (bf16_t)f0.y;  a[2]  = (bf16_t)f0.z;  a[3]  = (bf16_t)f0.w;
                a[4] = (bf16_t)f1.x;  a[5] = (bf16_t)f1.y;  a[6]  = (bf16_t)f1.z;  a[7]  = (bf16_t)f1.w;
                a[8] = (bf16_t)f2.x;  a[9] = (bf16_t)f2.y;  a[10] = (bf16_t)f2.z;  a[11] = (bf16_t)f2.w;
                a[12] = (bf16_t)f3.x; a[13] = (bf16_t)f3.y; a[14] = (bf16_t)f3.z;  a[15] = (bf16_t)f3.w;
            }
#pragma unroll
            for (int nt = 0; nt < NTC; ++nt) {
                const bf16_t* wp = &Wt[(nt * 16 + m16) * K + kk * 32 + hi * 8];
                union { v16bf v; v8bf h[2]; } bu;
                bu.h[0] = *(const v8bf*)(wp);
                bu.h[1] = *(const v8bf*)(wp + 16);
                acc[nt] = __builtin_amdgcn_wmma_f32_16x16x32_bf16(
                    false, a, false, bu.v, (short)0, acc[nt], false, false);
            }
        }

        // epilogue
#pragma unroll
        for (int nt = 0; nt < NTC; ++nt) {
            const int col  = nt * 16 + m16;
            const float bc = bias ? bias[col] : 0.0f;
#pragma unroll
            for (int r = 0; r < 8; ++r) {
                int m   = r + hi * 8;
                int row = rowBase + m;
                if (row < nrows) {
                    float v = acc[nt][r] + bc;
                    if (DO_SSP) v = sspf(v);
                    if (MODE == 0)      Y[(long long)row * NCOLS + col] = v;
                    else if (MODE == 1) Y[(long long)row * NCOLS + col] += v;
                    else                unsafeAtomicAdd(&Y[(long long)seg[row] * NCOLS + col], v);
                }
            }
        }
    }
}

// ---------------------------------------------------------- fused edge kernel
// per edge: d -> rbf[10] -> ssp(rbf@w1+b1) -> (WMMA) @w2+b2 -> *Cenv
//           -> * xj[src]  -> atomic scatter into agg[dst]
__global__ __launch_bounds__(256) void edge_kernel(
    const float* __restrict__ pos, const int* __restrict__ ei,
    const float* __restrict__ xj, float* __restrict__ agg,
    const float* __restrict__ w1, const float* __restrict__ b1,
    const float* __restrict__ w2, const float* __restrict__ b2,
    int E, int totalWaves)
{
    __shared__ __align__(16) bf16_t W2t[HD * HD];      // 32 KB: W2t[n*128+k] = w2[k][n]
    __shared__ float W1s[NGAUSS * HD];                 // 5 KB
    __shared__ float B1s[HD];
    __shared__ float RBF[8][16][NGAUSS];
    __shared__ float CEN[8][16];
    __shared__ int   SRCI[8][16];
    __shared__ int   DSTI[8][16];

    const int tid = threadIdx.x;
    for (int idx = tid; idx < HD * HD; idx += 256) {
        int k = idx / HD, n = idx % HD;
        W2t[n * HD + k] = (bf16_t)w2[idx];
    }
    for (int idx = tid; idx < NGAUSS * HD; idx += 256) W1s[idx] = w1[idx];
    if (tid < HD) B1s[tid] = b1[tid];
    __syncthreads();

    const int lane  = tid & 31;
    const int wave  = tid >> 5;
    const int gwave = blockIdx.x * 8 + wave;
    const int m16   = lane & 15;
    const int hi    = lane >> 4;

    const int chunks = (E + 15) >> 4;
    const int iters  = (chunks + totalWaves - 1) / totalWaves;
    const float delta = CUTOFF_F / (float)(NGAUSS - 1);
    const float coeff = -0.5f / (delta * delta);
    const float PI_F  = 3.14159265358979f;

    for (int it = 0; it < iters; ++it) {
        const int chunk = it * totalWaves + gwave;
        const long long eBase = (long long)chunk * 16;

        // ---- per-edge scalar phase (one edge per lane, lanes 0-15)
        if (lane < 16) {
            long long e = eBase + lane;
            bool valid = (chunk < chunks) && (e < (long long)E);
            int s = valid ? ei[e] : 0;
            int d = valid ? ei[(long long)E + e] : 0;
            float dx = pos[s * 3 + 0] - pos[d * 3 + 0];
            float dy = pos[s * 3 + 1] - pos[d * 3 + 1];
            float dz = pos[s * 3 + 2] - pos[d * 3 + 2];
            float dist = sqrtf(dx * dx + dy * dy + dz * dz);
            float cenv = valid ? 0.5f * (__cosf(dist * (PI_F / CUTOFF_F)) + 1.0f) : 0.0f;
            SRCI[wave][lane] = s;
            DSTI[wave][lane] = d;
            CEN[wave][lane]  = cenv;
#pragma unroll
            for (int g = 0; g < NGAUSS; ++g) {
                float t = dist - delta * (float)g;
                RBF[wave][lane][g] = valid ? __expf(coeff * t * t) : 0.0f;
            }
        }
        // same-wave LDS ordering is in-order; all lanes re-converge here

        // ---- filter MLP layer1 (VALU, K=10) straight into WMMA-A layout,
        //      then layer2 via bf16 WMMA against W2t (VALU co-executes with WMMA)
        float rb[NGAUSS];
#pragma unroll
        for (int g = 0; g < NGAUSS; ++g) rb[g] = RBF[wave][m16][g];

        v8f acc[8];
#pragma unroll
        for (int nt = 0; nt < 8; ++nt)
#pragma unroll
            for (int r = 0; r < 8; ++r) acc[nt][r] = 0.0f;

#pragma unroll
        for (int kk = 0; kk < 4; ++kk) {
            v16bf a;
#pragma unroll
            for (int j = 0; j < 8; ++j) {
                int col = kk * 32 + hi * 8 + j;
                float t0 = B1s[col];
#pragma unroll
                for (int g = 0; g < NGAUSS; ++g) t0 += rb[g] * W1s[g * HD + col];
                a[j] = (bf16_t)sspf(t0);
            }
#pragma unroll
            for (int j = 0; j < 8; ++j) {
                int col = kk * 32 + hi * 8 + 16 + j;
                float t0 = B1s[col];
#pragma unroll
                for (int g = 0; g < NGAUSS; ++g) t0 += rb[g] * W1s[g * HD + col];
                a[8 + j] = (bf16_t)sspf(t0);
            }
#pragma unroll
            for (int nt = 0; nt < 8; ++nt) {
                const bf16_t* wp = &W2t[(nt * 16 + m16) * HD + kk * 32 + hi * 8];
                union { v16bf v; v8bf h[2]; } bu;
                bu.h[0] = *(const v8bf*)(wp);
                bu.h[1] = *(const v8bf*)(wp + 16);
                acc[nt] = __builtin_amdgcn_wmma_f32_16x16x32_bf16(
                    false, a, false, bu.v, (short)0, acc[nt], false, false);
            }
        }

        // ---- epilogue: +b2, *cutoff, *xj[src], scatter-add agg[dst]
#pragma unroll
        for (int nt = 0; nt < 8; ++nt) {
            const int col  = nt * 16 + m16;
            const float bc = b2[col];
#pragma unroll
            for (int r = 0; r < 8; ++r) {
                int m = r + hi * 8;
                float Wv  = (acc[nt][r] + bc) * CEN[wave][m];
                int   s   = SRCI[wave][m];
                int   d   = DSTI[wave][m];
                float val = Wv * xj[(long long)s * HD + col];
                unsafeAtomicAdd(&agg[(long long)d * HD + col], val);
            }
        }
    }
}

// -------------------------------------------------------------- final linear
__global__ void pred_kernel(const float* __restrict__ pooled,
                            const float* __restrict__ pw,
                            const float* __restrict__ pb,
                            float* __restrict__ out)
{
    int tid = threadIdx.x;
    if (tid >= NBATCH * NTGT) return;
    int b = tid >> 2, t = tid & 3;
    float s = pb[t];
#pragma unroll 4
    for (int c = 0; c < HD; ++c) s += pooled[b * HD + c] * pw[c * NTGT + t];
    out[b * NTGT + t] = s;
}

// ------------------------------------------------------------------- driver
extern "C" void kernel_launch(void* const* d_in, const int* in_sizes, int n_in,
                              void* d_out, int out_size, void* d_ws, size_t ws_size,
                              hipStream_t stream)
{
    const int*   z      = (const int*)  d_in[0];
    const float* pos    = (const float*)d_in[1];
    const int*   batch  = (const int*)  d_in[2];
    const int*   ei     = (const int*)  d_in[3];
    const float* emb    = (const float*)d_in[4];
    const float* mlp_w1 = (const float*)d_in[5];
    const float* mlp_b1 = (const float*)d_in[6];
    const float* mlp_w2 = (const float*)d_in[7];
    const float* mlp_b2 = (const float*)d_in[8];
    const float* lin1_w = (const float*)d_in[9];
    const float* lin2_w = (const float*)d_in[10];
    const float* lin2_b = (const float*)d_in[11];
    const float* lin_w  = (const float*)d_in[12];
    const float* lin_b  = (const float*)d_in[13];
    const float* out1_w = (const float*)d_in[14];
    const float* out1_b = (const float*)d_in[15];
    const float* out2_w = (const float*)d_in[16];
    const float* out2_b = (const float*)d_in[17];
    const float* pred_w = (const float*)d_in[18];
    const float* pred_b = (const float*)d_in[19];
    float* out = (float*)d_out;

    const int N = in_sizes[0];
    const int E = in_sizes[3] / 2;

    float* h      = (float*)d_ws;
    float* tmp    = h   + (long long)N * HD;    // xj / hidden temporaries
    float* agg    = tmp + (long long)N * HD;
    float* pooled = agg + (long long)N * HD;

    const long long NH = (long long)N * HD;
    const int ZB = (int)((NH + 255) / 256);
    const int tiles  = (N + 15) / 16;
    const int mmBlk  = (tiles + 7) / 8;          // one 16-row tile per wave
    const int edgeBlk = 1250;                    // 10000 waves, grid-strided
    const int edgeWaves = edgeBlk * 8;

    // h = emb[z]
    init_h_kernel<<<ZB, 256, 0, stream>>>(h, emb, z, NH);

    for (int i = 0; i < 2; ++i) {
        // xj = h @ lin1  (no bias)
        wmma_matmul_kernel<HD, HD, 0, 0><<<mmBlk, 256, 0, stream>>>(
            h, lin1_w + (long long)i * HD * HD, nullptr, tmp, nullptr, N);
        // agg = 0
        zero_kernel<<<ZB, 256, 0, stream>>>(agg, NH);
        // fused edge pipeline: rbf -> MLP -> cutoff -> gather*modulate -> scatter
        edge_kernel<<<edgeBlk, 256, 0, stream>>>(
            pos, ei, tmp, agg,
            mlp_w1 + (long long)i * NGAUSS * HD, mlp_b1 + (long long)i * HD,
            mlp_w2 + (long long)i * HD * HD,     mlp_b2 + (long long)i * HD,
            E, edgeWaves);
        // tmp = ssp(agg @ lin2 + b)
        wmma_matmul_kernel<HD, HD, 1, 0><<<mmBlk, 256, 0, stream>>>(
            agg, lin2_w + (long long)i * HD * HD, lin2_b + (long long)i * HD,
            tmp, nullptr, N);
        // h += tmp @ lin + b
        wmma_matmul_kernel<HD, HD, 0, 1><<<mmBlk, 256, 0, stream>>>(
            tmp, lin_w + (long long)i * HD * HD, lin_b + (long long)i * HD,
            h, nullptr, N);
    }

    // tmp[:, :64] = ssp(h @ out1 + b)
    wmma_matmul_kernel<HD, HD / 2, 1, 0><<<mmBlk, 256, 0, stream>>>(
        h, out1_w, out1_b, tmp, nullptr, N);
    // pooled = segment_sum(tmp @ out2 + b, batch)
    zero_kernel<<<(NBATCH * HD + 255) / 256, 256, 0, stream>>>(pooled, NBATCH * HD);
    wmma_matmul_kernel<HD / 2, HD, 0, 2><<<mmBlk, 256, 0, stream>>>(
        tmp, out2_w, out2_b, pooled, batch, N);
    // out = pooled @ pred_w + pred_b
    pred_kernel<<<1, 256, 0, stream>>>(pooled, pred_w, pred_b, out);
}
